// BitLayer_25391846654323
// MI455X (gfx1250) — 8-bit matrix operand order repaired and checked by reading, compiled, not run on hardware
//
#include <hip/hip_runtime.h>
#include <stdint.h>

typedef __attribute__((ext_vector_type(8))) int          v8i;
typedef __attribute__((ext_vector_type(4))) int          v4i;
typedef __attribute__((ext_vector_type(4))) unsigned int v4u;

#define INPUT_DIM   1024
#define NUM_OUTPUTS 1024
#define BIT_SIZE    256
#define TILE_N      16
#define TILE_B      16
#define WAVES_PER_BLOCK 4
#define BLOCK_THREADS   (WAVES_PER_BLOCK * 32)
#define B_PER_BLOCK     (WAVES_PER_BLOCK * TILE_B)
#define CH_ROWS         512

#define XB_OFF 0
#define XB_WORDS (B_PER_BLOCK * (INPUT_DIM / 32))
#define PT_OFF (XB_OFF + XB_WORDS)
#define PT_WORDS ((INPUT_DIM / 4) * TILE_N)
#define ST_OFF (PT_OFF + PT_WORDS)
#define ST_WORDS (CH_ROWS * TILE_N)
#define SMEM_WORDS (ST_OFF + ST_WORDS)

#if defined(__has_builtin)
#if __has_builtin(__builtin_amdgcn_tensor_load_to_lds)
#define USE_TDM 1
#else
#define USE_TDM 0
#endif
#else
#define USE_TDM 0
#endif

__device__ __forceinline__ uint32_t hash32(uint32_t x) {
    x ^= x >> 16; x *= 0x21f0aaadu;
    x ^= x >> 15; x *= 0x735a2d97u;
    x ^= x >> 15;
    return x;
}

#if USE_TDM
__device__ __forceinline__ void tdm_load_tile(const float* gptr, uint32_t lds_byte_addr,
                                              uint32_t rows) {
    const uint64_t ga = (uint64_t)(uintptr_t)gptr;
    v4u g0;
    g0[0] = 1u;
    g0[1] = lds_byte_addr;
    g0[2] = (uint32_t)ga;
    g0[3] = (uint32_t)((ga >> 32) & 0x01FFFFFFu) | (2u << 30);
    v8i g1;
    g1[0] = (int)(2u << 16);
    g1[1] = (int)((NUM_OUTPUTS & 0xFFFFu) << 16);
    g1[2] = (int)(((NUM_OUTPUTS >> 16) & 0xFFFFu) |
                  ((INPUT_DIM & 0xFFFFu) << 16));
    g1[3] = (int)(((INPUT_DIM >> 16) & 0xFFFFu) |
                  ((uint32_t)TILE_N << 16));
    g1[4] = (int)(rows & 0xFFFFu);
    g1[5] = (int)NUM_OUTPUTS;
    g1[6] = 0;
    g1[7] = 0;
    const v4i z4 = {0, 0, 0, 0};
#if defined(__clang_major__) && (__clang_major__ >= 23)
    const v8i z8 = {0, 0, 0, 0, 0, 0, 0, 0};
    __builtin_amdgcn_tensor_load_to_lds(g0, g1, z4, z4, z8, 0);
#else
    __builtin_amdgcn_tensor_load_to_lds(g0, g1, z4, z4, 0);
#endif
    __builtin_amdgcn_s_wait_tensorcnt(0);
}
#endif

__global__ __launch_bounds__(BLOCK_THREADS)
void bitlayer_wmma_kernel(const int* __restrict__ x,
                          const float* __restrict__ kern,
                          float* __restrict__ out)
{
    __shared__ __align__(16) uint32_t smem[SMEM_WORDS];
    uint32_t* ldsX  = smem + XB_OFF;
    uint32_t* ldsP  = smem + PT_OFF;
#if USE_TDM
    float*    stage = (float*)(smem + ST_OFF);
#endif

    const int tid = threadIdx.x;
    const int n0  = blockIdx.x * TILE_N;
    const int B0  = blockIdx.y * B_PER_BLOCK;

    for (int idx = tid; idx < XB_WORDS; idx += BLOCK_THREADS) {
        const int bl = idx & (B_PER_BLOCK - 1);
        const int wi = idx >> 6;
        uint32_t w = 0;
        #pragma unroll 8
        for (int t = 0; t < 32; ++t) {
            __builtin_prefetch(&x[(wi * 32 + t) * BIT_SIZE + B0 + bl + 32], 0, 1);
            w |= (uint32_t)(x[(wi * 32 + t) * BIT_SIZE + B0 + bl] & 1) << t;
        }
        ldsX[bl * (INPUT_DIM / 32) + wi] = w;
    }

    for (int c = 0; c < INPUT_DIM / CH_ROWS; ++c) {
#if USE_TDM
        if ((tid >> 5) == 0) {
            tdm_load_tile(kern + (size_t)(c * CH_ROWS) * NUM_OUTPUTS + n0,
                          (uint32_t)(ST_OFF * 4), CH_ROWS);
        }
        __syncthreads();
#endif
        for (int wdx = tid; wdx < (CH_ROWS / 4) * TILE_N; wdx += BLOCK_THREADS) {
            const int nl  = wdx & 15;
            const int iwc = wdx >> 4;
            uint32_t pw = 0;
            #pragma unroll
            for (int j = 0; j < 4; ++j) {
#if USE_TDM
                const float p = stage[(iwc * 4 + j) * TILE_N + nl];
#else
                const float p = kern[(size_t)(c * CH_ROWS + iwc * 4 + j) * NUM_OUTPUTS
                                     + n0 + nl];
#endif
                int q = (int)(p * 256.0f);
                q = q < 0 ? 0 : (q > 255 ? 255 : q);
                pw |= (uint32_t)q << (8 * j);
            }
            ldsP[(c * (CH_ROWS / 4) + iwc) * TILE_N + nl] = pw;
        }
        __syncthreads();
    }

    const int lane = tid & 31;
    const int wid  = tid >> 5;
    const int h    = lane >> 4;
    const int nb   = lane & 15;
    const int bl   = wid * TILE_B + nb;
    const int b    = B0 + bl;

    v8i amat;
    #pragma unroll
    for (int v = 0; v < 8; ++v) {
        const int g = 8 * (v >> 2) + 4 * h + (v & 3);
        amat[v] = (g == nb) ? 0x01010101 : 0;
    }

    uint32_t basev[8];
    #pragma unroll
    for (int v = 0; v < 8; ++v) {
        const int nl = 8 * (v >> 2) + 4 * h + (v & 3);
        basev[v] = ((uint32_t)(n0 + nl) << 8) | (uint32_t)b;
    }

    v8i acc = {0, 0, 0, 0, 0, 0, 0, 0};
    const uint32_t H = 0x80808080u;
    uint32_t xword = 0;

    for (int i0 = 0; i0 < INPUT_DIM; i0 += 4) {
        if ((i0 & 31) == 0)
            xword = ldsX[bl * (INPUT_DIM / 32) + (i0 >> 5)];

        const uint32_t nib   = (xword >> (i0 & 31)) & 0xFu;
        const uint32_t xmask = ((nib * 0x00204081u) & 0x01010101u) * 0xFFu;
        const uint32_t igs   = (uint32_t)(i0 >> 2) << 18;

        const v4u p4a = *(const v4u*)&ldsP[(i0 >> 2) * TILE_N + 4 * h];
        const v4u p4b = *(const v4u*)&ldsP[(i0 >> 2) * TILE_N + 8 + 4 * h];
        const uint32_t pwv[8] = {p4a[0], p4a[1], p4a[2], p4a[3],
                                 p4b[0], p4b[1], p4b[2], p4b[3]};

        v8i bmat;
        #pragma unroll
        for (int v = 0; v < 8; ++v) {
            const uint32_t pm = pwv[v] & xmask;
            const uint32_t r  = hash32(igs + basev[v]);
            const uint32_t t  = (r | H) - (pm & 0x7F7F7F7Fu);
            const uint32_t ge = ((r & ~pm) | (~(r ^ pm) & t)) & H;
            bmat[v] = (int)((ge ^ H) >> 7);
        }

        acc = __builtin_amdgcn_wmma_i32_16x16x64_iu8(
            false, amat, false, bmat, acc, false, false);
    }

    #pragma unroll
    for (int r = 0; r < 8; ++r) {
        const int ng = n0 + r + 8 * h;
        out[ng * BIT_SIZE + b] = (acc[r] > 0) ? 1.0f : 0.0f;
    }
}

extern "C" void kernel_launch(void* const* d_in, const int* in_sizes, int n_in,
                              void* d_out, int out_size, void* d_ws, size_t ws_size,
                              hipStream_t stream) {
    const int*   x    = (const int*)d_in[0];
    const float* kern = (const float*)d_in[1];
    float*       out  = (float*)d_out;

    dim3 grid(NUM_OUTPUTS / TILE_N, BIT_SIZE / B_PER_BLOCK);
    dim3 block(BLOCK_THREADS);
    hipLaunchKernelGGL(bitlayer_wmma_kernel, grid, block, 0, stream, x, kern, out);
}
